// GHM_68143951118654
// MI455X (gfx1250) — matrix operand order repaired and checked by reading, not compiled, not run on hardware
//
#include <hip/hip_runtime.h>
#include <hip/hip_bf16.h>
#include <math.h>

// ---------------------------------------------------------------------------
// GHM forward for MI455X (gfx1250): bf16 WMMA (v_wmma_f32_16x16x32_bf16),
// wave32, 4 waves / 128-thread workgroups, LDS-tiled, deterministic
// (fixed-order partial reduction, no float atomics).
// Round 3: fix async-LDS builtin pointer types (int4 AS1 / int4 AS3).
// ---------------------------------------------------------------------------

#define DEVINL __device__ __forceinline__

typedef __bf16 bf16_t;
typedef __attribute__((ext_vector_type(4)))  bf16_t v4bf;
typedef __attribute__((ext_vector_type(8)))  bf16_t v8bf;
typedef __attribute__((ext_vector_type(16))) bf16_t v16bf;
typedef __attribute__((ext_vector_type(8)))  float  v8f;
typedef __attribute__((ext_vector_type(4)))  int    v4i;

static constexpr int NNODES = 200000;   // N
static constexpr int HDIM   = 256;      // H == IN
static constexpr int MDIM   = 64;       // M
static constexpr int CDIM   = 64;       // C
static constexpr int TM     = 64;       // node rows per workgroup tile
static constexpr int NTILES = NNODES / TM;         // 3125 (divides exactly)
static constexpr int LDA    = HDIM + 8;            // 264: Xs/Hs row stride (halves)
static constexpr int LTT    = TM + 8;              // 72: transposed-tile stride
static constexpr int LDU    = 2 * HDIM + 8;        // 520: concat buffer stride
static constexpr int GRIDP  = 512;                 // persistent WGs for phase A

// ---------------- async global<->LDS (guarded; falls back to sync copies) ---
#if __has_builtin(__builtin_amdgcn_global_load_async_to_lds_b128) && \
    __has_builtin(__builtin_amdgcn_s_wait_asynccnt)
#define ASYNC_LD 1
#else
#define ASYNC_LD 0
#endif
#if ASYNC_LD && __has_builtin(__builtin_amdgcn_global_store_async_from_lds_b128)
#define ASYNC_ST 1
#else
#define ASYNC_ST 0
#endif

typedef v4i __attribute__((address_space(1))) * gv4ip;   // global int4*
typedef v4i __attribute__((address_space(3))) * lv4ip;   // LDS int4*

// ---------------- branchless gelu ------------------------------------------
// tanh-approx gelu, algebraically folded:
//   gelu(x) ~= x * (1 - 1/(1 + exp2(x*(C1 + C2*x^2))))
// C1 = sqrt(2/pi)*2*log2(e), C2 = C1*0.044715.  Saturates correctly at +-inf.
DEVINL float fast_exp2(float x) {
#if __has_builtin(__builtin_amdgcn_exp2f)
    return __builtin_amdgcn_exp2f(x);
#else
    return exp2f(x);
#endif
}
DEVINL float fast_rcp(float x) {
#if __has_builtin(__builtin_amdgcn_rcpf)
    return __builtin_amdgcn_rcpf(x);
#else
    return 1.0f / x;
#endif
}
DEVINL float gelu_f(float x) {
    float x2  = x * x;
    float arg = x * fmaf(0.1029435f, x2, 2.3022080f);
    float r   = fast_rcp(1.0f + fast_exp2(arg));
    return x * (1.0f - r);
}

DEVINL v8f wmma_bf16(v16bf a, v16bf b, v8f c) {
    // D = A(16x32) * B(32x16) + C(16x16 f32)
    return __builtin_amdgcn_wmma_f32_16x16x32_bf16(
        false, a, false, b, (short)0, c, false, false);
}

// A fragment: 16x32 bf16, row-major source with stride `lda` halves.
// lane l: m=l&15, hi=l>>4; halves 0..7 <- K=k0+hi*8.. ; halves 8..15 <- K=k0+16+hi*8..
DEVINL v16bf frag_ldA(const bf16_t* __restrict__ base, int row0, int lda, int kTile) {
    const int l = threadIdx.x & 31;
    const int m = l & 15, hi = l >> 4;
    const bf16_t* p = base + (size_t)(row0 + m) * lda + kTile * 32 + hi * 8;
    v8bf lo = *(const v8bf*)p;
    v8bf hv = *(const v8bf*)(p + 16);
    return __builtin_shufflevector(lo, hv, 0,1,2,3,4,5,6,7,8,9,10,11,12,13,14,15);
}

// B fragment: 32x16 bf16, source is B^T row-major ([N][K], stride lda halves).
// lane l: n=l&15, hi=l>>4; halves 0..7 <- K=k0+hi*8.. ; halves 8..15 <- K=k0+16+hi*8..
DEVINL v16bf frag_ldB(const bf16_t* __restrict__ baseT, int col0, int lda, int kTile) {
    const int l = threadIdx.x & 31;
    const int n = l & 15, hi = l >> 4;
    const bf16_t* p = baseT + (size_t)(col0 + n) * lda + kTile * 32 + hi * 8;
    v8bf lo = *(const v8bf*)p;
    v8bf hv = *(const v8bf*)(p + 16);
    return __builtin_shufflevector(lo, hv, 0,1,2,3,4,5,6,7,8,9,10,11,12,13,14,15);
}

// ---------------------------------------------------------------------------
// K0: weight f32 -> bf16 transpose: out[c*R + r] = in[r*C + c]   (B^T layout)
// ---------------------------------------------------------------------------
__global__ void k_transpose(const float* __restrict__ in, bf16_t* __restrict__ outp,
                            int R, int Ccol) {
    int idx = blockIdx.x * blockDim.x + threadIdx.x;
    if (idx < R * Ccol) {
        int r = idx / Ccol, c = idx - r * Ccol;
        outp[(size_t)c * R + r] = (bf16_t)in[idx];
    }
}

// ---------------------------------------------------------------------------
// K1 (phase A): per node-tile:
//   Hf = gelu(X@W1 + b1); G = gelu(Hf@wA); Wt = G@wB
//   store Hf, Wt to global; accumulate partial P += Hf^T @ Wt  (256x64)
// Persistent grid-stride over tiles; P kept in WMMA accumulators.
// ---------------------------------------------------------------------------
template <bool IN_BF16>
__global__ __launch_bounds__(128)
void k_phaseA(const void* __restrict__ Xin,
              const bf16_t* __restrict__ w1t, const float* __restrict__ b1,
              const bf16_t* __restrict__ wAt, const bf16_t* __restrict__ wBt,
              bf16_t* __restrict__ HfW, bf16_t* __restrict__ WtW,
              float* __restrict__ partials, int nTiles) {
    __shared__ bf16_t Xs[TM * LDA];        // X tile (reused as G tile)
    __shared__ bf16_t Hs[TM * LDA];        // Hf tile, row-major (node, h)
    __shared__ bf16_t Hts[HDIM * LTT];     // Hf^T tile (h, node)
    __shared__ bf16_t Wtts[MDIM * LTT];    // Wt^T tile (m, node)

    const int tid = threadIdx.x;
    const int w   = tid >> 5;              // wave 0..3
    const int l   = tid & 31;
    const int n16 = l & 15;
    const int mhi = (l >> 4) * 8;
    const int r0  = w * 16;                // this wave's 16 rows within tile

    v8f pacc[4][4] = {};                   // persistent P tiles: (h-tile, m-tile)

    for (int t = blockIdx.x; t < nTiles; t += gridDim.x) {
        const int base = t * TM;
        __syncthreads();

        // ---- phase a: stage X tile into LDS as bf16 ----
        if (IN_BF16) {
            const bf16_t* Xb = (const bf16_t*)Xin;
#if ASYNC_LD
            for (int e = tid; e < TM * 32; e += 128) {
                int r = e >> 5, c = (e & 31) * 8;
                __builtin_amdgcn_global_load_async_to_lds_b128(
                    (gv4ip)(Xb + (size_t)(base + r) * HDIM + c),
                    (lv4ip)(Xs + r * LDA + c), 0, 0);
            }
            __builtin_amdgcn_s_wait_asynccnt(0);
#else
            for (int e = tid; e < TM * 32; e += 128) {
                int r = e >> 5, c = (e & 31) * 8;
                *(v8bf*)(Xs + r * LDA + c) =
                    *(const v8bf*)(Xb + (size_t)(base + r) * HDIM + c);
            }
#endif
        } else {
            const float* Xf = (const float*)Xin;
            for (int e = tid; e < TM * 64; e += 128) {
                int r = e >> 6, c = (e & 63) * 4;
                float4 v = *(const float4*)(Xf + (size_t)(base + r) * HDIM + c);
                v4bf o = { (bf16_t)v.x, (bf16_t)v.y, (bf16_t)v.z, (bf16_t)v.w };
                *(v4bf*)(Xs + r * LDA + c) = o;
            }
        }
        __syncthreads();

        // ---- phase b: Hf = gelu(X@W1 + b1) -> Hs (row-major) + Hts (transposed)
        {
            v16bf a[8];
#pragma unroll
            for (int kT = 0; kT < 8; ++kT) a[kT] = frag_ldA(Xs, r0, LDA, kT);
#pragma unroll
            for (int nT = 0; nT < 16; ++nT) {
                v8f acc = {};
#pragma unroll
                for (int kT = 0; kT < 8; ++kT)
                    acc = wmma_bf16(a[kT], frag_ldB(w1t, nT * 16, HDIM, kT), acc);
                float bias = b1[nT * 16 + n16];
                v8bf pk;
#pragma unroll
                for (int i = 0; i < 8; ++i) {
                    bf16_t h = (bf16_t)gelu_f(acc[i] + bias);
                    Hs[(size_t)(r0 + mhi + i) * LDA + nT * 16 + n16] = h;
                    pk[i] = h;
                }
                // packed transposed store: one ds_store_b128 per lane
                *(v8bf*)(Hts + (size_t)(nT * 16 + n16) * LTT + r0 + mhi) = pk;
            }
        }
        __syncthreads();

        // ---- coop: Hs -> HfW (global, bf16) ----
#if ASYNC_ST
        for (int e = tid; e < TM * 32; e += 128) {
            int r = e >> 5, c = (e & 31) * 8;
            __builtin_amdgcn_global_store_async_from_lds_b128(
                (gv4ip)(HfW + (size_t)(base + r) * HDIM + c),
                (lv4ip)(Hs + (size_t)r * LDA + c), 0, 0);
        }
#else
        for (int e = tid; e < TM * 32; e += 128) {
            int r = e >> 5, c = (e & 31) * 8;
            *(v8bf*)(HfW + (size_t)(base + r) * HDIM + c) =
                *(const v8bf*)(Hs + (size_t)r * LDA + c);
        }
#endif

        // ---- phase c: G = gelu(Hf@wA) -> Gs (reuses Xs) ----
        {
            v16bf a[8];
#pragma unroll
            for (int kT = 0; kT < 8; ++kT) a[kT] = frag_ldA(Hs, r0, LDA, kT);
#pragma unroll
            for (int nT = 0; nT < 16; ++nT) {
                v8f acc = {};
#pragma unroll
                for (int kT = 0; kT < 8; ++kT)
                    acc = wmma_bf16(a[kT], frag_ldB(wAt, nT * 16, HDIM, kT), acc);
#pragma unroll
                for (int i = 0; i < 8; ++i)
                    Xs[(size_t)(r0 + mhi + i) * LDA + nT * 16 + n16] =
                        (bf16_t)gelu_f(acc[i]);
            }
        }
        __syncthreads();

        // ---- phase d: Wt = G@wB -> WtW (global) + Wtts (transposed LDS) ----
        {
            v16bf a[8];
#pragma unroll
            for (int kT = 0; kT < 8; ++kT) a[kT] = frag_ldA(Xs, r0, LDA, kT);
#pragma unroll
            for (int nT = 0; nT < 4; ++nT) {
                v8f acc = {};
#pragma unroll
                for (int kT = 0; kT < 8; ++kT)
                    acc = wmma_bf16(a[kT], frag_ldB(wBt, nT * 16, HDIM, kT), acc);
                v8bf pk;
#pragma unroll
                for (int i = 0; i < 8; ++i) {
                    int node = base + r0 + mhi + i;
                    bf16_t h = (bf16_t)acc[i];
                    if (node < NNODES)
                        WtW[(size_t)node * MDIM + nT * 16 + n16] = h;
                    else
                        h = (bf16_t)0.0f;
                    pk[i] = h;
                }
                *(v8bf*)(Wtts + (size_t)(nT * 16 + n16) * LTT + r0 + mhi) = pk;
            }
        }
        __syncthreads();

        // ---- phase e: pacc += Hf^T(tile) @ Wt(tile), K = TM nodes ----
#pragma unroll
        for (int mi = 0; mi < 4; ++mi) {
            int mT = w * 4 + mi;           // h-tile 0..15 spread over 4 waves
            v16bf a0 = frag_ldA(Hts, mT * 16, LTT, 0);
            v16bf a1 = frag_ldA(Hts, mT * 16, LTT, 1);
#pragma unroll
            for (int nT = 0; nT < 4; ++nT) {
                pacc[mi][nT] = wmma_bf16(a0, frag_ldB(Wtts, nT * 16, LTT, 0), pacc[mi][nT]);
                pacc[mi][nT] = wmma_bf16(a1, frag_ldB(Wtts, nT * 16, LTT, 1), pacc[mi][nT]);
            }
        }

#if ASYNC_ST
        // Hs is re-written next iteration: make sure async stores consumed it.
        __builtin_amdgcn_s_wait_asynccnt(0);
#endif
    }

    // write this WG's partial P (256x64 f32); zero if it processed no tiles
    float* pout = partials + (size_t)blockIdx.x * (HDIM * MDIM);
#pragma unroll
    for (int mi = 0; mi < 4; ++mi)
#pragma unroll
        for (int nT = 0; nT < 4; ++nT)
#pragma unroll
            for (int i = 0; i < 8; ++i) {
                int h   = (w * 4 + mi) * 16 + mhi + i;
                int m64 = nT * 16 + n16;
                pout[(size_t)h * MDIM + m64] = pacc[mi][nT][i];
            }
}

// ---------------------------------------------------------------------------
// K2: agg = gelu(sum_p partials[p])  -> bf16 [256 h][64 m] (row-major)
// ---------------------------------------------------------------------------
__global__ __launch_bounds__(256)
void k_reduceP(const float* __restrict__ partials, bf16_t* __restrict__ aggB,
               int nParts) {
    int idx = blockIdx.x * 256 + threadIdx.x;   // 0..16383
    float s = 0.0f;
    for (int p = 0; p < nParts; ++p)
        s += partials[(size_t)p * (HDIM * MDIM) + idx];
    aggB[idx] = (bf16_t)gelu_f(s);
}

// ---------------------------------------------------------------------------
// K3 (phase C): per node-tile:
//   a = Wt_tile @ agg^T; u = gelu([a, Hf]); Xn = u @ W2 + b2
// ---------------------------------------------------------------------------
__global__ __launch_bounds__(128)
void k_phaseC(const bf16_t* __restrict__ HfW, const bf16_t* __restrict__ WtW,
              const bf16_t* __restrict__ aggB, const bf16_t* __restrict__ w2t,
              const float* __restrict__ b2, bf16_t* __restrict__ Xn) {
    __shared__ bf16_t us[TM * LDU];            // concat buffer (node, 512)

    const int tid = threadIdx.x;
    const int w   = tid >> 5;
    const int l   = tid & 31;
    const int n16 = l & 15;
    const int mhi = (l >> 4) * 8;
    const int r0  = w * 16;
    const int base = blockIdx.x * TM;

    // ---- phase 1: us[:,0:256] = gelu(Wt_tile @ agg^T) ----
    {
        const bf16_t* Wbase = WtW + (size_t)base * MDIM;
        v16bf a0 = frag_ldA(Wbase, r0, MDIM, 0);
        v16bf a1 = frag_ldA(Wbase, r0, MDIM, 1);
#pragma unroll
        for (int nT = 0; nT < 16; ++nT) {
            v8f acc = {};
            acc = wmma_bf16(a0, frag_ldB(aggB, nT * 16, MDIM, 0), acc);
            acc = wmma_bf16(a1, frag_ldB(aggB, nT * 16, MDIM, 1), acc);
#pragma unroll
            for (int i = 0; i < 8; ++i)
                us[(size_t)(r0 + mhi + i) * LDU + nT * 16 + n16] =
                    (bf16_t)gelu_f(acc[i]);
        }
    }

    // ---- phase 1b: us[:,256:512] = gelu(Hf); prefetch W2 while at it ----
    __builtin_prefetch(w2t + (size_t)tid * 1024, 0, 0);  // global_prefetch_b8
    for (int e = tid; e < TM * 32; e += 128) {
        int r = e >> 5, c = (e & 31) * 8;
        v8bf v = *(const v8bf*)(HfW + (size_t)(base + r) * HDIM + c);
        v8bf o;
#pragma unroll
        for (int i = 0; i < 8; ++i) o[i] = (bf16_t)gelu_f((float)v[i]);
        *(v8bf*)(us + (size_t)r * LDU + HDIM + c) = o;
    }
    __syncthreads();

    // ---- phase 2: Xn = u @ W2 + b2 ----
    {
        v16bf a[16];
#pragma unroll
        for (int kT = 0; kT < 16; ++kT) a[kT] = frag_ldA(us, r0, LDU, kT);
#pragma unroll
        for (int nT = 0; nT < 16; ++nT) {
            v8f acc = {};
#pragma unroll
            for (int kT = 0; kT < 16; ++kT)
                acc = wmma_bf16(a[kT], frag_ldB(w2t, nT * 16, 2 * HDIM, kT), acc);
            float bias = b2[nT * 16 + n16];
#pragma unroll
            for (int i = 0; i < 8; ++i) {
                int node = base + r0 + mhi + i;
                if (node < NNODES)
                    Xn[(size_t)node * HDIM + nT * 16 + n16] =
                        (bf16_t)(acc[i] + bias);
            }
        }
    }
}

// ---------------------------------------------------------------------------
// K4a: per-column partial sums of Xn over nodes
// ---------------------------------------------------------------------------
__global__ __launch_bounds__(256)
void k_colsum(const bf16_t* __restrict__ X, float* __restrict__ partialG) {
    int c = threadIdx.x, b = blockIdx.x;
    float s = 0.0f;
    for (int r = b; r < NNODES; r += 256)
        s += (float)X[(size_t)r * HDIM + c];
    partialG[(size_t)b * HDIM + c] = s;
}

// ---------------------------------------------------------------------------
// K4b: g = mean; out = g @ head_w + head_b
// ---------------------------------------------------------------------------
__global__ __launch_bounds__(256)
void k_head(const float* __restrict__ partialG, const bf16_t* __restrict__ headt,
            const float* __restrict__ hb, float* __restrict__ out) {
    __shared__ float gs[HDIM];
    int t = threadIdx.x;
    float s = 0.0f;
    for (int p = 0; p < 256; ++p) s += partialG[(size_t)p * HDIM + t];
    gs[t] = s * (1.0f / (float)NNODES);
    __syncthreads();
    if (t < CDIM) {
        float o = hb[t];
        for (int h = 0; h < HDIM; ++h)
            o += gs[h] * (float)headt[(size_t)t * HDIM + h];
        out[t] = o;
    }
}

// ---------------------------------------------------------------------------
// Host orchestration
// ---------------------------------------------------------------------------
extern "C" void kernel_launch(void* const* d_in, const int* in_sizes, int n_in,
                              void* d_out, int out_size, void* d_ws, size_t ws_size,
                              hipStream_t stream) {
    (void)in_sizes; (void)n_in; (void)out_size; (void)ws_size;

    const float* x  = (const float*)d_in[0];
    // d_in[1] = edge_index (unused: fully-connected graph task)
    const float* w1[2] = { (const float*)d_in[2],  (const float*)d_in[8]  };
    const float* b1[2] = { (const float*)d_in[3],  (const float*)d_in[9]  };
    const float* wA[2] = { (const float*)d_in[4],  (const float*)d_in[10] };
    const float* wB[2] = { (const float*)d_in[5],  (const float*)d_in[11] };
    const float* w2[2] = { (const float*)d_in[6],  (const float*)d_in[12] };
    const float* b2[2] = { (const float*)d_in[7],  (const float*)d_in[13] };
    const float* hw = (const float*)d_in[14];
    const float* hb = (const float*)d_in[15];
    float* out = (float*)d_out;

    // carve workspace (deterministic layout, 256B aligned)
    char* p = (char*)d_ws;
    auto carve = [&](size_t bytes) -> char* {
        char* r = p;
        p += (bytes + 255) & ~(size_t)255;
        return r;
    };
    bf16_t* w1t[2], *wAt[2], *wBt[2], *w2t[2];
    for (int b = 0; b < 2; ++b) {
        w1t[b] = (bf16_t*)carve((size_t)HDIM * HDIM * 2);
        wAt[b] = (bf16_t*)carve((size_t)HDIM * HDIM * 2);
        wBt[b] = (bf16_t*)carve((size_t)MDIM * HDIM * 2);
        w2t[b] = (bf16_t*)carve((size_t)HDIM * 2 * HDIM * 2);
    }
    bf16_t* headt    = (bf16_t*)carve((size_t)CDIM * HDIM * 2);
    bf16_t* aggB     = (bf16_t*)carve((size_t)HDIM * MDIM * 2);
    float*  partialG = (float*) carve((size_t)256 * HDIM * 4);
    float*  partials = (float*) carve((size_t)GRIDP * HDIM * MDIM * 4);
    bf16_t* HfW      = (bf16_t*)carve((size_t)NNODES * HDIM * 2);
    bf16_t* WtW      = (bf16_t*)carve((size_t)NNODES * MDIM * 2);
    bf16_t* XnW      = (bf16_t*)carve((size_t)NNODES * HDIM * 2);

    auto T = [&](const float* src, bf16_t* dst, int R, int C) {
        int total = R * C;
        k_transpose<<<(total + 255) / 256, 256, 0, stream>>>(src, dst, R, C);
    };
    for (int b = 0; b < 2; ++b) {
        T(w1[b], w1t[b], HDIM, HDIM);          // (IN,H)  -> [H][IN]
        T(wA[b], wAt[b], HDIM, HDIM);          // (H,H)   -> [H][H]
        T(wB[b], wBt[b], HDIM, MDIM);          // (H,M)   -> [M][H]
        T(w2[b], w2t[b], 2 * HDIM, HDIM);      // (2H,H)  -> [H][2H]
    }
    T(hw, headt, HDIM, CDIM);                  // (H,C)   -> [C][H]

    for (int b = 0; b < 2; ++b) {
        if (b == 0)
            k_phaseA<false><<<GRIDP, 128, 0, stream>>>(
                (const void*)x, w1t[0], b1[0], wAt[0], wBt[0],
                HfW, WtW, partials, NTILES);
        else
            k_phaseA<true><<<GRIDP, 128, 0, stream>>>(
                (const void*)XnW, w1t[1], b1[1], wAt[1], wBt[1],
                HfW, WtW, partials, NTILES);

        k_reduceP<<<(HDIM * MDIM) / 256, 256, 0, stream>>>(partials, aggB, GRIDP);

        k_phaseC<<<NTILES, 128, 0, stream>>>(HfW, WtW, aggB, w2t[b], b2[b], XnW);
    }

    k_colsum<<<256, 256, 0, stream>>>(XnW, partialG);
    k_head<<<1, 256, 0, stream>>>(partialG, headt, hb, out);
}
